// LLamaSelfAttention_7799660610256
// MI455X (gfx1250) — compile-verified, matrix operand order checked by reading, not run on hardware
//
#include <hip/hip_runtime.h>
#include <hip/hip_bf16.h>

// ---------------------------------------------------------------------------
// Problem constants (reference: B=2, T=2048, C=4096, H=32, D=128)
// ---------------------------------------------------------------------------
constexpr int Bc = 2;
constexpr int Tc = 2048;
constexpr int Cc = 4096;
constexpr int Hc = 32;
constexpr int Dc = 128;

typedef __attribute__((ext_vector_type(16))) __bf16 v16bf;
typedef __attribute__((ext_vector_type(8)))  __bf16 v8bf;
typedef __attribute__((ext_vector_type(8)))  float  v8f;
typedef __attribute__((ext_vector_type(4)))  unsigned v4u;
typedef __attribute__((ext_vector_type(8)))  unsigned v8u;

union FragU { v16bf v; v8bf h[2]; };

// fp32 -> bf16, round-to-nearest-even (bit manipulation: no reliance on cast lowering)
static __device__ inline __bf16 f2bf(float f) {
    unsigned u = __builtin_bit_cast(unsigned, f);
    unsigned r = (u + 0x7FFFu + ((u >> 16) & 1u)) >> 16;
    unsigned short h = (unsigned short)r;
    return __builtin_bit_cast(__bf16, h);
}
static __device__ inline float bf2f(__bf16 b) {
    unsigned short h = __builtin_bit_cast(unsigned short, b);
    unsigned u = ((unsigned)h) << 16;
    return __builtin_bit_cast(float, u);
}

static __device__ inline v8f wmma_bf16(v16bf a, v16bf b, v8f c) {
    return __builtin_amdgcn_wmma_f32_16x16x32_bf16(
        /*neg_a=*/false, a, /*neg_b=*/false, b,
        /*c_mod=*/(short)0, c, /*reuse_a=*/false, /*reuse_b=*/false);
}

// Load one 16x32 bf16 fragment (A or B operand). `row` points at this lane's
// K-contiguous 32-element row slice. CDNA5 16-bit operand layout: lanes 0-15
// hold K {0..7,16..23}, lanes 16-31 hold K {8..15,24..31}.
static __device__ inline v16bf ld_frag(const __bf16* row, int hi) {
    FragU u;
    u.h[0] = *(const v8bf*)(row + hi * 8);
    u.h[1] = *(const v8bf*)(row + 16 + hi * 8);
    return u.v;
}

// Generic pointer to a __shared__ object: low 32 bits are the LDS byte offset.
static __device__ inline unsigned lds_off(const void* p) {
    return (unsigned)(uintptr_t)p;
}

// ---------------------------------------------------------------------------
// Tensor Data Mover: async-load a 2D tile of 2-byte elements into LDS.
//   tile_x : contiguous elements per row (== tensor_dim0, no OOB)
//   tile_y : number of rows            (== tensor_dim1)
//   stride : row stride in ELEMENTS (tensor_dim0_stride)
// D# built per CDNA5 ISA 8.3/8.4; issued via inline asm (portable across the
// 5-arg/6-arg builtin arity split). Tracked by TENSORcnt.
// ---------------------------------------------------------------------------
static __device__ inline void tdm_load_2d_b16(unsigned lds_addr, const void* gptr,
                                              unsigned tile_x, unsigned tile_y,
                                              unsigned stride) {
    unsigned long long ga = (unsigned long long)(uintptr_t)gptr;
    v4u g0;
    g0.x = 1u;                                        // count=1, user mode
    g0.y = lds_addr;                                  // lds_addr [63:32]
    g0.z = (unsigned)ga;                              // global_addr [95:64]
    g0.w = (unsigned)((ga >> 32) & 0x1FFFFFFu)        // global_addr [120:96]
         | (2u << 30);                                // type=2 ("image")
    v8u g1;
    g1.s0 = (1u << 16);                               // data_size=1 -> 2 bytes
    g1.s1 = (tile_x & 0xFFFFu) << 16;                 // tensor_dim0[15:0]
    g1.s2 = (tile_x >> 16) | ((tile_y & 0xFFFFu) << 16); // dim0 hi | dim1 lo
    g1.s3 = (tile_y >> 16) | (tile_x << 16);          // dim1 hi | tile_dim0
    g1.s4 = (tile_y & 0xFFFFu);                       // tile_dim1 (tile_dim2=0)
    g1.s5 = stride;                                   // tensor_dim0_stride lo32
    g1.s6 = 0u;                                       // stride hi | dim1_stride lo
    g1.s7 = 0u;
    asm volatile("tensor_load_to_lds %0, %1" :: "s"(g0), "s"(g1) : "memory");
}

// ---------------------------------------------------------------------------
// fp32 -> bf16 convert
// ---------------------------------------------------------------------------
__global__ __launch_bounds__(256) void convert_bf16_kernel(
    const float* __restrict__ in, __bf16* __restrict__ out, size_t n) {
    size_t i = (size_t)blockIdx.x * blockDim.x + threadIdx.x;
    if (i < n) out[i] = f2bf(in[i]);
}

// fp32 (n x n row-major) -> bf16 transposed (so every GEMM operand is K-contiguous)
__global__ __launch_bounds__(256) void transpose_convert_kernel(
    const float* __restrict__ in, __bf16* __restrict__ out, int n) {
    size_t idx = (size_t)blockIdx.x * blockDim.x + threadIdx.x;
    if (idx >= (size_t)n * n) return;
    int i = (int)(idx / n);
    int j = (int)(idx % n);
    out[(size_t)j * n + i] = f2bf(in[idx]);
}

// ---------------------------------------------------------------------------
// GEMM: out[M,N] = A[M,K] * Bt[N,K]^T   (both operands bf16, K-contiguous)
// 256 threads = 8 waves; block tile 128x128; wave tile 32x64 (2x4 WMMA).
// K-step 32; LDS tiles are double-buffered and filled by the Tensor Data
// Mover: wave 0 issues tensor_load_to_lds for tile kb+1, then waits
// TENSORcnt<=2 (in-order => tile kb's pair has landed) and barriers. The DMA
// for the next tile runs fully behind the 8-WMMA burst of the current one.
// ---------------------------------------------------------------------------
template <bool OUTF32>
__global__ __launch_bounds__(256) void gemm_bt_kernel(
    const __bf16* __restrict__ A, const __bf16* __restrict__ Bt,
    void* __restrict__ outp, int M, int N, int K) {
    __shared__ __bf16 As[2][128 * 32];
    __shared__ __bf16 Bs[2][128 * 32];

    const int tid  = threadIdx.x;
    const int lane = tid & 31;
    const int wave = tid >> 5;
    const int lc   = lane & 15;
    const int hi   = lane >> 4;

    const int m0 = blockIdx.y * 128;
    const int n0 = blockIdx.x * 128;
    const int wm = (wave >> 1) * 32;   // 0..96
    const int wn = (wave & 1) * 64;    // 0 or 64

    // scalar (SGPR) condition: exactly wave 0 issues the TDM descriptors
    const bool issuer = (__builtin_amdgcn_readfirstlane(tid) == 0);

    const __bf16* Abase = &A[(size_t)m0 * K];
    const __bf16* Bbase = &Bt[(size_t)n0 * K];

    v8f acc[2][4] = {};
    const int nkb = K / 32;

    if (issuer) {   // prologue: DMA K-tile 0 into buffer 0
        tdm_load_2d_b16(lds_off(&As[0][0]), Abase, 32u, 128u, (unsigned)K);
        tdm_load_2d_b16(lds_off(&Bs[0][0]), Bbase, 32u, 128u, (unsigned)K);
    }

    for (int kb = 0; kb < nkb; ++kb) {
        const int cur   = kb & 1;
        const bool more = (kb + 1) < nkb;

        if (issuer) {
            if (more) {   // DMA next tile into the other buffer
                tdm_load_2d_b16(lds_off(&As[cur ^ 1][0]),
                                Abase + (size_t)(kb + 1) * 32, 32u, 128u, (unsigned)K);
                tdm_load_2d_b16(lds_off(&Bs[cur ^ 1][0]),
                                Bbase + (size_t)(kb + 1) * 32, 32u, 128u, (unsigned)K);
                __builtin_amdgcn_s_wait_tensorcnt(2);  // current tile's pair done
            } else {
                __builtin_amdgcn_s_wait_tensorcnt(0);
            }
        }
        __syncthreads();

        v16bf af[2], bfr[4];
#pragma unroll
        for (int i = 0; i < 2; ++i)
            af[i] = ld_frag(&As[cur][(wm + i * 16 + lc) * 32], hi);
#pragma unroll
        for (int j = 0; j < 4; ++j)
            bfr[j] = ld_frag(&Bs[cur][(wn + j * 16 + lc) * 32], hi);
#pragma unroll
        for (int i = 0; i < 2; ++i)
#pragma unroll
            for (int j = 0; j < 4; ++j)
                acc[i][j] = wmma_bf16(af[i], bfr[j], acc[i][j]);

        __syncthreads();   // all reads of buf[cur] done before TDM may refill it
    }

    // C/D layout: VGPR r holds (M = r + 8*hi, N = lc) of each 16x16 tile.
#pragma unroll
    for (int i = 0; i < 2; ++i) {
#pragma unroll
        for (int j = 0; j < 4; ++j) {
#pragma unroll
            for (int r = 0; r < 8; ++r) {
                int mm = m0 + wm + i * 16 + r + 8 * hi;
                int nn = n0 + wn + j * 16 + lc;
                float val = acc[i][j][r];
                if constexpr (OUTF32)
                    ((float*)outp)[(size_t)mm * N + nn] = val;
                else
                    ((__bf16*)outp)[(size_t)mm * N + nn] = f2bf(val);
            }
        }
    }
}

// ---------------------------------------------------------------------------
// RoPE applied in place to Q and K (bf16 [B*T, C], head h at cols h*D..)
// ---------------------------------------------------------------------------
__global__ __launch_bounds__(256) void rope_kernel(
    __bf16* __restrict__ Qb, __bf16* __restrict__ Kb) {
    int idx = blockIdx.x * blockDim.x + threadIdx.x;  // B*T*H*(D/2) threads
    int d    = idx & (Dc / 2 - 1);
    int rest = idx >> 6;                // / (D/2)
    int h    = rest & (Hc - 1);
    int bt   = rest >> 5;               // / H
    if (bt >= Bc * Tc) return;
    int t = bt & (Tc - 1);

    float freq = __powf(10000.0f, -((float)(2 * d)) / (float)Dc);
    float ang  = (float)t * freq;
    float s, c;
    __sincosf(ang, &s, &c);

    size_t base = (size_t)bt * Cc + h * Dc;
    {
        float x1 = bf2f(Qb[base + d]);
        float x2 = bf2f(Qb[base + d + Dc / 2]);
        Qb[base + d]          = f2bf(x1 * c - x2 * s);
        Qb[base + d + Dc / 2] = f2bf(x2 * c + x1 * s);
    }
    {
        float x1 = bf2f(Kb[base + d]);
        float x2 = bf2f(Kb[base + d + Dc / 2]);
        Kb[base + d]          = f2bf(x1 * c - x2 * s);
        Kb[base + d + Dc / 2] = f2bf(x2 * c + x1 * s);
    }
}

// ---------------------------------------------------------------------------
// Flash-style causal attention. 128 threads = 4 waves per block; block owns
// 64 query rows (16 per wave). Per k-block of 32 keys: the K-tile (32x128,
// row stride C) is DMA'd into LDS by the Tensor Data Mover while all threads
// stage the transposed V-tile manually (TDM can't transpose); one
// s_wait_tensorcnt + barrier publishes both. K/V are L2-resident anyway
// (67 MB << 192 MB L2). Per-wave causal guard is wave-uniform so EXEC is
// all-ones at every WMMA.
// ---------------------------------------------------------------------------
__global__ __launch_bounds__(128) void attn_kernel(
    const __bf16* __restrict__ Qb, const __bf16* __restrict__ Kb,
    const __bf16* __restrict__ Vb, __bf16* __restrict__ Ob) {
    __shared__ __bf16 Ks[32 * 128];      // K rows (k-contig d)       8 KB
    __shared__ __bf16 Vt[128 * 32];      // V transposed [d][k]       8 KB
    __shared__ __bf16 Pb[4][16 * 32];    // per-wave P tiles          4 KB

    const int tid  = threadIdx.x;
    const int lane = tid & 31;
    const int wave = tid >> 5;
    const int lc   = lane & 15;
    const int hi   = lane >> 4;

    const int qb0 = blockIdx.x * 64;       // block's first query row
    const int qw  = qb0 + wave * 16;       // this wave's first query row
    const int h   = blockIdx.y;
    const int b   = blockIdx.z;

    const __bf16* Qh = Qb + (size_t)b * Tc * Cc + h * Dc;
    const __bf16* Kh = Kb + (size_t)b * Tc * Cc + h * Dc;
    const __bf16* Vh = Vb + (size_t)b * Tc * Cc + h * Dc;

    const bool issuer = (__builtin_amdgcn_readfirstlane(tid) == 0);

    // Q fragments for this wave's 16 rows, all of D=128 (4 x 16x32 frags)
    v16bf qf[4];
#pragma unroll
    for (int kk = 0; kk < 4; ++kk)
        qf[kk] = ld_frag(Qh + (size_t)(qw + lc) * Cc + kk * 32, hi);

    float m[8], l[8];
    v8f o[8] = {};
#pragma unroll
    for (int r = 0; r < 8; ++r) { m[r] = -3.0e38f; l[r] = 0.0f; }

    const float sc = 0.08838834764831845f;       // 1/sqrt(128)
    const int nkb_blk  = (qb0 + 64 + 31) / 32;   // k-blocks any wave needs
    const int nkb_wave = (qw + 16 + 31) / 32;    // k-blocks this wave needs

    // manual V staging map (transpose): thread covers V row vkv, d-chunk vdc
    const int vkv = tid & 31;              // 0..31 (V row within k-block)
    const int vdc = (tid >> 5) * 32;       // 0,32,64,96 (d chunk)

    for (int kb = 0; kb < nkb_blk; ++kb) {
        const int k0 = kb * 32;

        // ---- TDM: K tile (32 rows x 128 d, row stride C) -> Ks ----
        if (issuer)
            tdm_load_2d_b16(lds_off(&Ks[0]), Kh + (size_t)k0 * Cc,
                            128u, 32u, (unsigned)Cc);

        // ---- manual transposed V staging (overlaps the K DMA) ----
        {
            const __bf16* vrow = Vh + (size_t)(k0 + vkv) * Cc + vdc;
            union { v8bf v[4]; __bf16 e[32]; } tv;
#pragma unroll
            for (int q = 0; q < 4; ++q) tv.v[q] = *(const v8bf*)(vrow + q * 8);
#pragma unroll
            for (int i = 0; i < 32; ++i) Vt[(vdc + i) * 32 + vkv] = tv.e[i];
        }
        if (issuer) __builtin_amdgcn_s_wait_tensorcnt(0);
        __syncthreads();

        if (kb < nkb_wave) {   // wave-uniform causal guard
            // ---- S = Q K^T (16 x 32): preload all 8 K fragments, then burst ----
            v16bf kf0[4], kf1[4];
#pragma unroll
            for (int kk = 0; kk < 4; ++kk) {
                kf0[kk] = ld_frag(&Ks[lc * 128 + kk * 32], hi);
                kf1[kk] = ld_frag(&Ks[(lc + 16) * 128 + kk * 32], hi);
            }
            v8f s0 = {}, s1 = {};
#pragma unroll
            for (int kk = 0; kk < 4; ++kk) {
                s0 = wmma_bf16(qf[kk], kf0[kk], s0);
                s1 = wmma_bf16(qf[kk], kf1[kk], s1);
            }

            // ---- scale + causal mask + online softmax (fully unrolled) ----
            float alpha[8];
#pragma unroll
            for (int r = 0; r < 8; ++r) {
                int q_idx = qw + r + 8 * hi;
                float a0 = s0[r] * sc + (((k0 + lc) > q_idx) ? -1.0e9f : 0.0f);
                float a1 = s1[r] * sc + (((k0 + 16 + lc) > q_idx) ? -1.0e9f : 0.0f);
                float loc = fmaxf(a0, a1);
#pragma unroll
                for (int off = 1; off < 16; off <<= 1)
                    loc = fmaxf(loc, __shfl_xor(loc, off, 32));
                float mnew = fmaxf(m[r], loc);
                alpha[r] = __expf(m[r] - mnew);
                float p0 = __expf(a0 - mnew);
                float p1 = __expf(a1 - mnew);
                float ps = p0 + p1;
#pragma unroll
                for (int off = 1; off < 16; off <<= 1)
                    ps += __shfl_xor(ps, off, 32);
                l[r] = l[r] * alpha[r] + ps;
                m[r] = mnew;
                Pb[wave][(r + 8 * hi) * 32 + lc]      = f2bf(p0);
                Pb[wave][(r + 8 * hi) * 32 + lc + 16] = f2bf(p1);
            }
#pragma unroll
            for (int j = 0; j < 8; ++j)
#pragma unroll
                for (int r = 0; r < 8; ++r)
                    o[j][r] = o[j][r] * alpha[r];

            // ---- O += P V (16 x 128): 8 WMMAs against shared Vt ----
            v16bf pa = ld_frag(&Pb[wave][lc * 32], hi);
            v16bf vf[8];
#pragma unroll
            for (int j = 0; j < 8; ++j)
                vf[j] = ld_frag(&Vt[(j * 16 + lc) * 32], hi);
#pragma unroll
            for (int j = 0; j < 8; ++j)
                o[j] = wmma_bf16(pa, vf[j], o[j]);
        }
        __syncthreads();
    }

    // ---- normalize and write (bf16, [B*T, C] layout for the Wo GEMM) ----
    __bf16* obase = Ob + ((size_t)b * Tc + qw) * Cc + h * Dc;
#pragma unroll
    for (int r = 0; r < 8; ++r) {
        float inv = 1.0f / l[r];
        int M = r + 8 * hi;
#pragma unroll
        for (int j = 0; j < 8; ++j)
            obase[(size_t)M * Cc + j * 16 + lc] = f2bf(o[j][r] * inv);
    }
}

// ---------------------------------------------------------------------------
// Host launcher
// ---------------------------------------------------------------------------
extern "C" void kernel_launch(void* const* d_in, const int* in_sizes, int n_in,
                              void* d_out, int out_size, void* d_ws, size_t ws_size,
                              hipStream_t stream) {
    const float* X  = (const float*)d_in[0];
    const float* Wq = (const float*)d_in[1];
    const float* Wk = (const float*)d_in[2];
    const float* Wv = (const float*)d_in[3];
    const float* Wo = (const float*)d_in[4];
    // d_in[5] = attention_mask (causal mask recomputed on device)
    float* out = (float*)d_out;

    const size_t SX = (size_t)Bc * Tc * Cc;  // activation elems
    const size_t SW = (size_t)Cc * Cc;       // weight elems

    char* ws = (char*)d_ws;
    __bf16* Xb  = (__bf16*)ws;                 ws += SX * sizeof(__bf16);
    __bf16* Wqt = (__bf16*)ws;                 ws += SW * sizeof(__bf16);
    __bf16* Wkt = (__bf16*)ws;                 ws += SW * sizeof(__bf16);
    __bf16* Wvt = (__bf16*)ws;                 ws += SW * sizeof(__bf16);
    __bf16* Wot = (__bf16*)ws;                 ws += SW * sizeof(__bf16);
    __bf16* Qb  = (__bf16*)ws;                 ws += SX * sizeof(__bf16);
    __bf16* Kbf = (__bf16*)ws;                 ws += SX * sizeof(__bf16);
    __bf16* Vbf = (__bf16*)ws;                 ws += SX * sizeof(__bf16);
    __bf16* AOb = (__bf16*)ws;                 ws += SX * sizeof(__bf16);

    // 1) down-convert activations + transpose-convert weights to bf16
    convert_bf16_kernel<<<(unsigned)((SX + 255) / 256), 256, 0, stream>>>(X, Xb, SX);
    const unsigned wgrid = (unsigned)((SW + 255) / 256);
    transpose_convert_kernel<<<wgrid, 256, 0, stream>>>(Wq, Wqt, Cc);
    transpose_convert_kernel<<<wgrid, 256, 0, stream>>>(Wk, Wkt, Cc);
    transpose_convert_kernel<<<wgrid, 256, 0, stream>>>(Wv, Wvt, Cc);
    transpose_convert_kernel<<<wgrid, 256, 0, stream>>>(Wo, Wot, Cc);

    // 2) Q/K/V projections (bf16 WMMA GEMMs, 128x128 block tiles, TDM staging)
    const int Mrows = Bc * Tc;
    dim3 ggrid(Cc / 128, Mrows / 128);
    gemm_bt_kernel<false><<<ggrid, 256, 0, stream>>>(Xb, Wqt, Qb, Mrows, Cc, Cc);
    gemm_bt_kernel<false><<<ggrid, 256, 0, stream>>>(Xb, Wkt, Kbf, Mrows, Cc, Cc);
    gemm_bt_kernel<false><<<ggrid, 256, 0, stream>>>(Xb, Wvt, Vbf, Mrows, Cc, Cc);

    // 3) RoPE on Q and K
    const unsigned nrope = (unsigned)((size_t)Bc * Tc * Hc * (Dc / 2));
    rope_kernel<<<nrope / 256, 256, 0, stream>>>(Qb, Kbf);

    // 4) causal flash attention (4 waves / 64 q-rows per block, TDM K-tiles)
    dim3 agrid(Tc / 64, Hc, Bc);
    attn_kernel<<<agrid, 128, 0, stream>>>(Qb, Kbf, Vbf, AOb);

    // 5) output projection -> fp32 d_out
    gemm_bt_kernel<true><<<ggrid, 256, 0, stream>>>(AOb, Wot, out, Mrows, Cc, Cc);
}
